// MultiHeadAttention_31138512896521
// MI455X (gfx1250) — matrix operand order repaired and checked by reading, not compiled, not run on hardware
//
#include <hip/hip_runtime.h>
#include <hip/hip_bf16.h>

// ---------------------------------------------------------------------------
// MHA for MI455X (gfx1250): bf16 WMMA (f32 accum), flash-attention online
// softmax, async global->LDS K/V staging (ASYNCcnt), LDS P-tile relayout,
// wave32 shuffles, 4-wide N register blocking in the projection GEMMs.
// ---------------------------------------------------------------------------

#define BATCH  2
#define SEQ    2048
#define DMODEL 2048
#define HEADS  16
#define DK     128            // DMODEL / HEADS

typedef __attribute__((ext_vector_type(16))) __bf16 v16bf;
typedef __attribute__((ext_vector_type(8)))  float  v8f;

// ---- bf16 helpers (bit-level, RNE) ----------------------------------------
__device__ __forceinline__ __bf16 bits2bf(unsigned short u) {
    union { unsigned short u; __bf16 b; } x; x.u = u; return x.b;
}
__device__ __forceinline__ unsigned short f2bf_bits(float f) {
    union { float f; unsigned u; } x; x.f = f;
    unsigned r = x.u + 0x7FFFu + ((x.u >> 16) & 1u);   // round to nearest even
    return (unsigned short)(r >> 16);
}
__device__ __forceinline__ __bf16 f2bf(float f) { return bits2bf(f2bf_bits(f)); }

__device__ __forceinline__ v8f wmma_bf16(v16bf a, v16bf b, v8f c) {
    // D = A(16x32 bf16) * B(32x16 bf16) + C(16x16 f32)
    return __builtin_amdgcn_wmma_f32_16x16x32_bf16(
        /*neg_a=*/false, a, /*neg_b=*/false, b,
        /*c_mod=*/(short)0, c, /*reuse_a=*/false, /*reuse_b=*/false);
}

__device__ __forceinline__ float redmax16(float v) {
#pragma unroll
    for (int off = 1; off < 16; off <<= 1) v = fmaxf(v, __shfl_xor(v, off, 32));
    return v;
}
__device__ __forceinline__ float redsum16(float v) {
#pragma unroll
    for (int off = 1; off < 16; off <<= 1) v += __shfl_xor(v, off, 32);
    return v;
}

// Truncate generic (flat) pointer to the 32-bit LDS byte offset
// (flat LDS aperture: offset is the low 32 address bits).
__device__ __forceinline__ unsigned lds_off(const void* p) {
    return (unsigned)(unsigned long long)p;
}

// ---------------------------------------------------------------------------
// GEMM:  Y = X @ W^T + bias      (X: MxK,  W: NxK row-major,  Y: MxN)
//   A_BF16   : X is bf16 (bits) else fp32 (converted to bf16 on the fly)
//   PERM_OUT : write bf16 into [B,H,S,DK] head-major layout, else fp32 row-major
// One wave computes a 16x64 strip (4 N-tiles), reusing the A fragment 4x.
// ---------------------------------------------------------------------------
template<bool A_BF16, bool PERM_OUT>
__global__ __launch_bounds__(256) void mha_gemm_kernel(
    const void* __restrict__ Aptr, const float* __restrict__ W,
    const float* __restrict__ bias, void* __restrict__ outp)
{
    constexpr int M  = BATCH * SEQ;   // 4096
    constexpr int N  = DMODEL;        // 2048
    constexpr int K  = DMODEL;        // 2048
    constexpr int NB = 4;             // N-tiles per wave
    constexpr int tilesNB = (N / 16) / NB;   // 32

    const int lane  = threadIdx.x & 31;
    const int wave  = blockIdx.x * (blockDim.x >> 5) + (threadIdx.x >> 5);
    const int mtile = wave / tilesNB;
    const int ntb   = wave % tilesNB;
    if (mtile * 16 >= M) return;

    const int n16  = lane & 15;
    const int half = lane >> 4;
    const int arow = mtile * 16 + n16;            // A-fragment: M = lane%16

    int ncol[NB];
#pragma unroll
    for (int j = 0; j < NB; ++j) ncol[j] = (ntb * NB + j) * 16 + n16;

    v8f acc[NB];
#pragma unroll
    for (int j = 0; j < NB; ++j)
#pragma unroll
        for (int i = 0; i < 8; ++i) acc[j][i] = 0.0f;

    const float*          Af = (const float*)Aptr;
    const unsigned short* Ab = (const unsigned short*)Aptr;

    for (int k0 = 0; k0 < K; k0 += 32) {
        // ---- A fragment: lane holds row arow; K = {half*8..+7, 16+half*8..+7}
        v16bf a;
        if constexpr (A_BF16) {
            const unsigned short* rp = Ab + (size_t)arow * K + k0;
#pragma unroll
            for (int e = 0; e < 16; ++e) {
                int kk = (e < 8) ? (half * 8 + e) : (16 + half * 8 + (e - 8));
                a[e] = bits2bf(rp[kk]);
            }
        } else {
            const float* rp = Af + (size_t)arow * K + k0;
            __builtin_prefetch(rp + 32, 0, 0);
            float4 f0 = *(const float4*)(rp + half * 8);
            float4 f1 = *(const float4*)(rp + half * 8 + 4);
            float4 f2 = *(const float4*)(rp + 16 + half * 8);
            float4 f3 = *(const float4*)(rp + 16 + half * 8 + 4);
            a[0]  = f2bf(f0.x); a[1]  = f2bf(f0.y); a[2]  = f2bf(f0.z); a[3]  = f2bf(f0.w);
            a[4]  = f2bf(f1.x); a[5]  = f2bf(f1.y); a[6]  = f2bf(f1.z); a[7]  = f2bf(f1.w);
            a[8]  = f2bf(f2.x); a[9]  = f2bf(f2.y); a[10] = f2bf(f2.z); a[11] = f2bf(f2.w);
            a[12] = f2bf(f3.x); a[13] = f2bf(f3.y); a[14] = f2bf(f3.z); a[15] = f2bf(f3.w);
        }

        // ---- 4 B fragments: B[k][n] = W[ncol][k]; lane reads its W row,
        //      K = {half*8..+7, 16+half*8..+7} (matches A). A reused 4x.
#pragma unroll
        for (int j = 0; j < NB; ++j) {
            const float* wr = W + (size_t)ncol[j] * K + k0 + half * 8;
            float4 w0 = *(const float4*)(wr + 0);
            float4 w1 = *(const float4*)(wr + 4);
            float4 w2 = *(const float4*)(wr + 16);
            float4 w3 = *(const float4*)(wr + 20);
            v16bf bm;
            bm[0]  = f2bf(w0.x); bm[1]  = f2bf(w0.y); bm[2]  = f2bf(w0.z); bm[3]  = f2bf(w0.w);
            bm[4]  = f2bf(w1.x); bm[5]  = f2bf(w1.y); bm[6]  = f2bf(w1.z); bm[7]  = f2bf(w1.w);
            bm[8]  = f2bf(w2.x); bm[9]  = f2bf(w2.y); bm[10] = f2bf(w2.z); bm[11] = f2bf(w2.w);
            bm[12] = f2bf(w3.x); bm[13] = f2bf(w3.y); bm[14] = f2bf(w3.z); bm[15] = f2bf(w3.w);
            acc[j] = wmma_bf16(a, bm, acc[j]);
        }
    }

#pragma unroll
    for (int j = 0; j < NB; ++j) {
        const float bvv = bias[ncol[j]];
#pragma unroll
        for (int r = 0; r < 8; ++r) {
            float yv = acc[j][r] + bvv;            // C/D: M = r + 8*half, N = ncol
            int grow = mtile * 16 + r + 8 * half;
            if constexpr (PERM_OUT) {
                int bb = grow / SEQ, ss = grow % SEQ;
                int hh = ncol[j] / DK, dk = ncol[j] % DK;
                ((unsigned short*)outp)[(((size_t)bb * HEADS + hh) * SEQ + ss) * DK + dk]
                    = f2bf_bits(yv);
            } else {
                ((float*)outp)[(size_t)grow * DMODEL + ncol[j]] = yv;
            }
        }
    }
}

// ---------------------------------------------------------------------------
// Flash attention. 4 waves/block; QT=128 is divisible by 4, so all waves in a
// block share one (b,h) and stream identical K/V tiles: each 32x128 bf16 tile
// (8 KB, contiguous in head-major layout) is staged into LDS once per block
// via GLOBAL_LOAD_ASYNC_TO_LDS_B128 (ASYNCcnt), then shared by all 4 waves.
// ---------------------------------------------------------------------------
__global__ __launch_bounds__(128) void mha_attn_kernel(
    const unsigned short* __restrict__ Qh, const unsigned short* __restrict__ Kh,
    const unsigned short* __restrict__ Vh, const int* __restrict__ mask,
    unsigned short* __restrict__ concat)
{
    __shared__ unsigned short lds_k[32 * DK];      // 8 KB  K tile (row-major)
    __shared__ unsigned short lds_v[32 * DK];      // 8 KB  V tile (row-major)
    __shared__ unsigned short lds_p[4][16 * 32];   // 4 KB  per-wave P staging

    const int lane    = threadIdx.x & 31;
    const int waveBlk = threadIdx.x >> 5;
    const int wg      = blockIdx.x * 4 + waveBlk;

    constexpr int QT = SEQ / 16;          // 128 query tiles per (b,h)
    const int qtile = wg % QT;
    const int bh    = wg / QT;            // identical for all 4 waves in block
    const int h     = bh % HEADS;
    const int b     = bh / HEADS;

    const int n16  = lane & 15;
    const int half = lane >> 4;

    const size_t headBase = ((size_t)(b * HEADS + h)) * SEQ * DK;
    const unsigned short* Qb = Qh + headBase;
    const unsigned short* Kb = Kh + headBase;
    const unsigned short* Vb = Vh + headBase;

    const unsigned ldsk_base = lds_off(&lds_k[0]);
    const unsigned ldsv_base = lds_off(&lds_v[0]);

    // ---- Q fragments: 16x128 bf16 = 4 chunks of 16x32 (A layout) ----
    v16bf qfrag[4];
    const int qrow = qtile * 16 + n16;
#pragma unroll
    for (int c = 0; c < 4; ++c) {
        const unsigned short* rp = Qb + (size_t)qrow * DK + c * 32;
#pragma unroll
        for (int e = 0; e < 16; ++e) {
            int kk = half * 16 + e;
            qfrag[c][e] = bits2bf(rp[kk]);
        }
    }

    v8f ofrag[8];
#pragma unroll
    for (int nt = 0; nt < 8; ++nt)
#pragma unroll
        for (int i = 0; i < 8; ++i) ofrag[nt][i] = 0.0f;

    float run_m[8], run_l[8];
#pragma unroll
    for (int r = 0; r < 8; ++r) { run_m[r] = -3.0e38f; run_l[r] = 0.0f; }

    const float scale = 0.08838834764831845f;   // 1/sqrt(DK)

    for (int jb = 0; jb < SEQ / 32; ++jb) {
        // ---- async-stage K/V tiles (each 8 KB contiguous) into LDS ----
        // 512 x 16B chunks per tile; 128 threads issue 4+4 async b128 copies.
        {
            const char* Kt = (const char*)(Kb + (size_t)(jb * 32) * DK);
            const char* Vt = (const char*)(Vb + (size_t)(jb * 32) * DK);
#pragma unroll
            for (int t = 0; t < 4; ++t) {
                int c = (int)threadIdx.x + t * 128;          // chunk id 0..511
                unsigned loff = (unsigned)(c * 16);
                unsigned lk = ldsk_base + loff;
                unsigned lv = ldsv_base + loff;
                unsigned long long gk = (unsigned long long)(Kt + c * 16);
                unsigned long long gv = (unsigned long long)(Vt + c * 16);
                asm volatile("global_load_async_to_lds_b128 %0, %1, off"
                             :: "v"(lk), "v"(gk) : "memory");
                asm volatile("global_load_async_to_lds_b128 %0, %1, off"
                             :: "v"(lv), "v"(gv) : "memory");
            }
            asm volatile("s_wait_asynccnt 0" ::: "memory");
        }
        __syncthreads();   // K/V tile visible to all 4 waves

        // ---- scores S = Q Kt for 32 keys: two 16x16 tiles (K from LDS) ----
        v8f sfrag[2];
#pragma unroll
        for (int t = 0; t < 2; ++t) {
#pragma unroll
            for (int i = 0; i < 8; ++i) sfrag[t][i] = 0.0f;
            const int krow = t * 16 + n16;        // B-frag: key index = N lane
#pragma unroll
            for (int c = 0; c < 4; ++c) {
                const unsigned short* rp = &lds_k[krow * DK + c * 32 + half * 16];
                v16bf kf;
#pragma unroll
                for (int e = 0; e < 16; ++e) kf[e] = bits2bf(rp[e]);
                sfrag[t] = wmma_bf16(qfrag[c], kf, sfrag[t]);
            }
        }

        // ---- scale + mask ----
#pragma unroll
        for (int t = 0; t < 2; ++t) {
#pragma unroll
            for (int r = 0; r < 8; ++r) {
                float sv = sfrag[t][r] * scale;
                int qr = qtile * 16 + r + 8 * half;
                int kc = jb * 32 + t * 16 + n16;
                if (mask[((size_t)b * SEQ + qr) * SEQ + kc] == 0) sv = -1.0e9f;
                sfrag[t][r] = sv;
            }
        }

        // ---- online softmax update ----
        float corr[8];
#pragma unroll
        for (int r = 0; r < 8; ++r) {
            float bm = redmax16(fmaxf(sfrag[0][r], sfrag[1][r]));
            float nm = fmaxf(run_m[r], bm);
            corr[r]  = __expf(run_m[r] - nm);
            run_m[r] = nm;
        }
#pragma unroll
        for (int r = 0; r < 8; ++r) {
            float p0 = __expf(sfrag[0][r] - run_m[r]);
            float p1 = __expf(sfrag[1][r] - run_m[r]);
            sfrag[0][r] = p0; sfrag[1][r] = p1;
            run_l[r] = run_l[r] * corr[r] + redsum16(p0 + p1);
        }
#pragma unroll
        for (int nt = 0; nt < 8; ++nt)
#pragma unroll
            for (int r = 0; r < 8; ++r) ofrag[nt][r] *= corr[r];

        // ---- P: C-layout -> LDS -> A-layout (16x32 bf16, per-wave slice) ----
        unsigned short* pl = lds_p[waveBlk];
#pragma unroll
        for (int t = 0; t < 2; ++t)
#pragma unroll
            for (int r = 0; r < 8; ++r)
                pl[(r + 8 * half) * 32 + t * 16 + n16] = f2bf_bits(sfrag[t][r]);
        asm volatile("s_wait_dscnt 0" ::: "memory");   // stores land before reload
        v16bf pf;
#pragma unroll
        for (int e = 0; e < 16; ++e) {
            int kk = half * 16 + e;
            pf[e] = bits2bf(pl[n16 * 32 + kk]);
        }

        // ---- O += P @ V  (8 output column tiles of 16; V from LDS) ----
#pragma unroll
        for (int nt = 0; nt < 8; ++nt) {
            v16bf vf;
#pragma unroll
            for (int e = 0; e < 16; ++e) {
                int kr = half * 16 + e;               // B-frag: K = half*16 + e
                vf[e] = bits2bf(lds_v[kr * DK + nt * 16 + n16]);
            }
            ofrag[nt] = wmma_bf16(pf, vf, ofrag[nt]);
        }

        __syncthreads();   // all waves done with this K/V tile before overwrite
    }

    // ---- normalize and write concat[b, s, h*DK + d] (bf16) ----
    float inv[8];
#pragma unroll
    for (int r = 0; r < 8; ++r) inv[r] = 1.0f / run_l[r];
#pragma unroll
    for (int nt = 0; nt < 8; ++nt) {
#pragma unroll
        for (int r = 0; r < 8; ++r) {
            float o = ofrag[nt][r] * inv[r];
            int qr  = qtile * 16 + r + 8 * half;
            int col = h * DK + nt * 16 + n16;
            concat[((size_t)b * SEQ + qr) * DMODEL + col] = f2bf_bits(o);
        }
    }
}

// ---------------------------------------------------------------------------
extern "C" void kernel_launch(void* const* d_in, const int* in_sizes, int n_in,
                              void* d_out, int out_size, void* d_ws, size_t ws_size,
                              hipStream_t stream) {
    const float* q    = (const float*)d_in[0];
    const float* k    = (const float*)d_in[1];
    const float* v    = (const float*)d_in[2];
    const int*   mask = (const int*)d_in[3];
    const float* Wq   = (const float*)d_in[4];
    const float* bq   = (const float*)d_in[5];
    const float* Wk   = (const float*)d_in[6];
    const float* bk   = (const float*)d_in[7];
    const float* Wv   = (const float*)d_in[8];
    const float* bv   = (const float*)d_in[9];
    const float* Wo   = (const float*)d_in[10];
    const float* bo   = (const float*)d_in[11];

    // workspace: Qh | Kh | Vh | concat, each B*S*D bf16 (16 MiB) -> 64 MiB
    const size_t elems = (size_t)BATCH * SEQ * DMODEL;
    unsigned short* Qh = (unsigned short*)d_ws;
    unsigned short* Kh = Qh + elems;
    unsigned short* Vh = Kh + elems;
    unsigned short* Cc = Vh + elems;

    constexpr int gemmWaves = (BATCH * SEQ / 16) * (DMODEL / 16) / 4; // 8192
    dim3 gblk(256), ggrid(gemmWaves / 8);                             // 8 waves/block

    mha_gemm_kernel<false, true><<<ggrid, gblk, 0, stream>>>(q, Wq, bq, Qh);
    mha_gemm_kernel<false, true><<<ggrid, gblk, 0, stream>>>(k, Wk, bk, Kh);
    mha_gemm_kernel<false, true><<<ggrid, gblk, 0, stream>>>(v, Wv, bv, Vh);

    constexpr int attnWaves = BATCH * HEADS * (SEQ / 16);             // 4096
    mha_attn_kernel<<<attnWaves / 4, 128, 0, stream>>>(Qh, Kh, Vh, mask, Cc);

    mha_gemm_kernel<true, false><<<ggrid, gblk, 0, stream>>>(Cc, Wo, bo, d_out);
}
